// Attention_82746839924863
// MI455X (gfx1250) — matrix operand order repaired and checked by reading, not compiled, not run on hardware
//
#include <hip/hip_runtime.h>
#include <math.h>

typedef __attribute__((ext_vector_type(16))) __bf16 v16bf;
typedef __attribute__((ext_vector_type(8)))  float  v8f;

#define B_     16
#define H_     32
#define KVH_   8
#define D_     128
#define BS_    16
#define MAXB_  256
#define REP_   4
#define WAVES_ 4
#define SPLIT_ 4                       // KV-range splits across grid.z (flash-decoding)
#define KVSTRIDE (KVH_ * D_)           // floats between consecutive tokens in cache
#define SCALE_ 0.08838834764831845f    // 1/sqrt(128)

__device__ __forceinline__ v8f zero8() {
  v8f z;
  #pragma unroll
  for (int i = 0; i < 8; ++i) z[i] = 0.f;
  return z;
}
__device__ __forceinline__ v16bf cvt_bf16(const float* f) {
  v16bf r;
  #pragma unroll
  for (int e = 0; e < 16; ++e) r[e] = (__bf16)f[e];
  return r;
}
// 16 floats: two contiguous 8-float runs at p and p+16 (A-operand K-tile rows)
__device__ __forceinline__ void load_k16(const float* p, float4* dst) {
  dst[0] = ((const float4*)p)[0];
  dst[1] = ((const float4*)p)[1];
  dst[2] = ((const float4*)(p + 16))[0];
  dst[3] = ((const float4*)(p + 16))[1];
}
// 16 token-strided floats (B-operand V gather; coalesced across lanes per element)
__device__ __forceinline__ void load_v16(const float* p, ptrdiff_t dblk, float* dst) {
  #pragma unroll
  for (int e = 0; e < 16; ++e) dst[e] = (e < 8) ? p[e * KVSTRIDE] : p[dblk + (e - 8) * KVSTRIDE];
}

// ---------------- Pass 1: per-(kv-head, batch, split) partial attention ----------------
__launch_bounds__(128)
__global__ void attn_partial(const float* __restrict__ q,
                             const float* __restrict__ k_cache,
                             const float* __restrict__ v_cache,
                             const int*   __restrict__ block_tables,
                             const int*   __restrict__ ctx_lens,
                             float* __restrict__ part_m,
                             float* __restrict__ part_l,
                             float* __restrict__ part_o) {
  const int g     = blockIdx.x;
  const int b     = blockIdx.y;
  const int split = blockIdx.z;
  const int tid   = threadIdx.x;
  const int wave  = tid >> 5;
  const int lane  = tid & 31;
  const int ln16  = lane & 15;
  const int hi    = lane >> 4;

  __shared__ float q_lds[REP_][D_];
  __shared__ float red_m[WAVES_][REP_];
  __shared__ float red_l[WAVES_][REP_];
  __shared__ float red_o[WAVES_][REP_][D_];

  const int n_hist = ctx_lens[b] - 1;           // cache tokens; new token handled in pass 2
  const int* btab  = block_tables + b * MAXB_;

  const float* qbase = q + ((size_t)b * H_ + g * REP_) * D_;
  for (int i = tid; i < REP_ * D_; i += 128)
    q_lds[i / D_][i % D_] = qbase[i] * SCALE_;
  __syncthreads();

  // B-operand: Q^T per 32-d chunk (column = head ln16, element e -> d = c*32 + hi*8 + e (+8 if e>=8))
  v16bf bq[4];
  #pragma unroll
  for (int c = 0; c < 4; ++c)
    #pragma unroll
    for (int e = 0; e < 16; ++e) {
      float f = (ln16 < REP_) ? q_lds[ln16][c * 32 + hi * 8 + e + (e >= 8 ? 8 : 0)] : 0.f;
      bq[c][e] = (__bf16)f;
    }

  // per-lane online-softmax state for head ln16 (lanes with ln16>=4 carry benign junk)
  float m_h = -INFINITY, l_h = 0.f;
  v8f acc[8];
  #pragma unroll
  for (int dd = 0; dd < 8; ++dd) acc[dd] = zero8();

  const int n_tiles = (n_hist + 31) >> 5;       // 32 tokens (2 cache blocks) per tile
  for (int t = split * WAVES_ + wave; t < n_tiles; t += WAVES_ * SPLIT_) {
    const int tok0 = t << 5;
    const int blk0 = btab[2 * t];
    const int blk1 = btab[2 * t + 1];

    // ---- S^T = K · Q^T : A = K-tile rows (token ln16), double-buffered ----
    const float* kr0 = k_cache + ((size_t)blk0 * BS_ + ln16) * KVSTRIDE + g * D_ + hi * 8;
    const float* kr1 = k_cache + ((size_t)blk1 * BS_ + ln16) * KVSTRIDE + g * D_ + hi * 8;
    float4 kb[2][4];
    load_k16(kr0, kb[0]);
    v8f sv0 = zero8(), sv1 = zero8();
    #pragma unroll
    for (int idx = 0; idx < 8; ++idx) {        // idx = gq*4 + c
      if (idx < 7) {
        const int nx = idx + 1;
        load_k16((nx < 4 ? kr0 : kr1) + (nx & 3) * 32, kb[nx & 1]);
      }
      v16bf ak = cvt_bf16((const float*)kb[idx & 1]);
      if (idx < 4)
        sv0 = __builtin_amdgcn_wmma_f32_16x16x32_bf16(false, ak, false, bq[idx & 3],
                                                      (short)0, sv0, false, false);
      else
        sv1 = __builtin_amdgcn_wmma_f32_16x16x32_bf16(false, ak, false, bq[idx & 3],
                                                      (short)0, sv1, false, false);
    }

    // ---- in-register softmax: lane holds head ln16, tokens hi*8+i (+16 for sv1) ----
    float lmax = -INFINITY;
    #pragma unroll
    for (int i = 0; i < 8; ++i) {
      float x0 = (tok0 + hi * 8 + i      < n_hist) ? sv0[i] : -INFINITY;
      float x1 = (tok0 + 16 + hi * 8 + i < n_hist) ? sv1[i] : -INFINITY;
      sv0[i] = x0; sv1[i] = x1;
      lmax = fmaxf(lmax, fmaxf(x0, x1));
    }
    lmax = fmaxf(lmax, __shfl_xor(lmax, 16, 32));   // combine lane pair (h, h+16)
    const float mnew  = fmaxf(m_h, lmax);
    const float alpha = __expf(m_h - mnew);
    float lsum = 0.f;
    v16bf ap;                                       // P already in A-operand layout!
    #pragma unroll
    for (int i = 0; i < 8; ++i) {
      float p0 = __expf(sv0[i] - mnew);
      float p1 = __expf(sv1[i] - mnew);
      lsum += p0 + p1;
      ap[i]     = (__bf16)p0;
      ap[i + 8] = (__bf16)p1;
    }
    lsum += __shfl_xor(lsum, 16, 32);
    l_h = l_h * alpha + lsum;
    m_h = mnew;

    // rescale accumulator: element r (lanes<16) is head r -> broadcast alphas
    const float a0 = __shfl(alpha, 0, 32);
    const float a1 = __shfl(alpha, 1, 32);
    const float a2 = __shfl(alpha, 2, 32);
    const float a3 = __shfl(alpha, 3, 32);
    #pragma unroll
    for (int dd = 0; dd < 8; ++dd) {
      acc[dd][0] *= a0; acc[dd][1] *= a1; acc[dd][2] *= a2; acc[dd][3] *= a3;
    }

    // ---- O += P · V : B = V gather (token-strided), double-buffered ----
    const float* vbase = v_cache + ((size_t)blk0 * BS_ + hi * 8) * KVSTRIDE
                       + g * D_ + ln16;
    const ptrdiff_t dblk = ((ptrdiff_t)blk1 - blk0) * BS_ * KVSTRIDE;
    float vf[2][16];
    load_v16(vbase, dblk, vf[0]);
    #pragma unroll
    for (int dd = 0; dd < 8; ++dd) {
      if (dd < 7) load_v16(vbase + (dd + 1) * 16, dblk, vf[(dd + 1) & 1]);
      v16bf bv = cvt_bf16(vf[dd & 1]);
      acc[dd] = __builtin_amdgcn_wmma_f32_16x16x32_bf16(false, ap, false, bv,
                                                        (short)0, acc[dd], false, false);
    }
  }

  // ---- per-wave partials -> LDS ----
  if (!hi) {
    #pragma unroll
    for (int r = 0; r < REP_; ++r)
      #pragma unroll
      for (int dd = 0; dd < 8; ++dd)
        red_o[wave][r][dd * 16 + ln16] = acc[dd][r];
    if (ln16 < REP_) { red_m[wave][ln16] = m_h; red_l[wave][ln16] = l_h; }
  }
  __syncthreads();

  // ---- workgroup-internal combine -> unnormalized partial in workspace ----
  const int pidx = (b * KVH_ + g) * SPLIT_ + split;
  #pragma unroll
  for (int r = 0; r < REP_; ++r) {
    float ms = red_m[0][r];
    #pragma unroll
    for (int w = 1; w < WAVES_; ++w) ms = fmaxf(ms, red_m[w][r]);
    float ls = 0.f, os = 0.f;
    #pragma unroll
    for (int w = 0; w < WAVES_; ++w) {
      float a = __expf(red_m[w][r] - ms);
      ls += red_l[w][r] * a;
      os += red_o[w][r][tid] * a;
    }
    part_o[((size_t)pidx * REP_ + r) * D_ + tid] = os;
    if (tid == 0) { part_m[pidx * REP_ + r] = ms; part_l[pidx * REP_ + r] = ls; }
  }
}

// ---------------- Pass 2: combine splits + fold in the NEW token ----------------
__launch_bounds__(128)
__global__ void attn_combine(const float* __restrict__ q,
                             const float* __restrict__ k_new,
                             const float* __restrict__ v_new,
                             const float* __restrict__ part_m,
                             const float* __restrict__ part_l,
                             const float* __restrict__ part_o,
                             float* __restrict__ out) {
  const int g    = blockIdx.x;
  const int b    = blockIdx.y;
  const int tid  = threadIdx.x;
  const int wave = tid >> 5;
  const int lane = tid & 31;

  __shared__ float q_lds[REP_][D_];
  __shared__ float s_last[REP_];

  const float* qbase = q + ((size_t)b * H_ + g * REP_) * D_;
  for (int i = tid; i < REP_ * D_; i += 128)
    q_lds[i / D_][i % D_] = qbase[i] * SCALE_;
  __syncthreads();

  // new token's score (replaces the reference's cache write): wave w does head w
  {
    const float* knew = k_new + ((size_t)b * KVH_ + g) * D_;
    float part = 0.f;
    for (int d = lane; d < D_; d += 32) part += q_lds[wave][d] * knew[d];
    #pragma unroll
    for (int off = 16; off > 0; off >>= 1) part += __shfl_xor(part, off, 32);
    if (lane == 0) s_last[wave] = part;
  }
  __syncthreads();

  const float vn = v_new[((size_t)b * KVH_ + g) * D_ + tid];
  const int pbase = (b * KVH_ + g) * SPLIT_;
  #pragma unroll
  for (int r = 0; r < REP_; ++r) {
    float ms = s_last[r];
    #pragma unroll
    for (int s = 0; s < SPLIT_; ++s) ms = fmaxf(ms, part_m[(pbase + s) * REP_ + r]);
    float lsum = 0.f, osum = 0.f;
    #pragma unroll
    for (int s = 0; s < SPLIT_; ++s) {
      float a = __expf(part_m[(pbase + s) * REP_ + r] - ms);
      lsum += part_l[(pbase + s) * REP_ + r] * a;
      osum += part_o[((size_t)(pbase + s) * REP_ + r) * D_ + tid] * a;
    }
    float pl = __expf(s_last[r] - ms);
    lsum += pl;
    osum += pl * vn;
    out[((size_t)b * H_ + g * REP_ + r) * D_ + tid] = osum / lsum;
  }
}

extern "C" void kernel_launch(void* const* d_in, const int* in_sizes, int n_in,
                              void* d_out, int out_size, void* d_ws, size_t ws_size,
                              hipStream_t stream) {
  (void)in_sizes; (void)n_in; (void)out_size; (void)ws_size;
  const float* q  = (const float*)d_in[0];
  const float* k  = (const float*)d_in[1];
  const float* v  = (const float*)d_in[2];
  const float* kc = (const float*)d_in[3];
  const float* vc = (const float*)d_in[4];
  const int* bt   = (const int*)d_in[5];
  const int* cl   = (const int*)d_in[6];
  // d_in[7] (slot_mapping) unused: the new token's K/V are folded analytically
  // into the softmax instead of mutating the cache inputs.
  float* out = (float*)d_out;

  const int nmat = B_ * KVH_ * SPLIT_ * REP_;     // 2048
  float* part_m = (float*)d_ws;
  float* part_l = part_m + nmat;
  float* part_o = part_l + nmat;                  // nmat * D_ floats (~1 MB)

  dim3 g1(KVH_, B_, SPLIT_);
  attn_partial<<<g1, 128, 0, stream>>>(q, kc, vc, bt, cl, part_m, part_l, part_o);
  dim3 g2(KVH_, B_);
  attn_combine<<<g2, 128, 0, stream>>>(q, k, v, part_m, part_l, part_o, out);
}
